// LogicEncoder_89489938579990
// MI455X (gfx1250) — matrix operand order repaired and checked by reading, not compiled, not run on hardware
//
#include <hip/hip_runtime.h>
#include <hip/hip_bf16.h>
#include <stdint.h>

// ---------------------------------------------------------------------------
// TreeLSTM encoder for MI455X (gfx1250, wave32).
// GEMMs via V_WMMA_F32_16X16X32_BF16 (f32 accumulate). Cell state c kept f32.
// In-place tree layout: level-d node j lives at slot (j << (d+1)).
// Levels d=0..5 (m>=32): full-tile WMMA kernel, global c/h.
// Levels d=6..10 (m<=16): fused single kernel, one block per tree, c/h in LDS.
// ---------------------------------------------------------------------------

#define LATENT 128
#define NTREE  64
#define NLEAF  2048

typedef __attribute__((ext_vector_type(16))) __bf16 v16bf;
typedef __attribute__((ext_vector_type(8)))  float  v8f;

union FragBF { uint4 q[2]; v16bf v; };

__device__ __forceinline__ unsigned short f2bf(float f) {
  unsigned int u = __float_as_uint(f);
  unsigned int r = (u + 0x7FFFu + ((u >> 16) & 1u)) >> 16;   // RNE
  return (unsigned short)r;
}

__device__ __forceinline__ float sigm(float x) { return 1.0f / (1.0f + __expf(-x)); }

__device__ __forceinline__ v8f wmma_bf16(v16bf a, v16bf b, v8f c) {
  // 8 args: (neg_a, A, neg_b, B, c_mod, C, reuse_a, reuse_b)
  return __builtin_amdgcn_wmma_f32_16x16x32_bf16(false, a, false, b, (short)0, c,
                                                 false, false);
}

// ---------------------------------------------------------------------------
// Prep: pack B matrices into WMMA-fragment-swizzled bf16 layout + fold biases.
// Level B logical (384 x 512): rows 0:128=xe-side, 128:256=lh, 256:384=rh;
// col blocks: [0:128)=gate i, [128:256)=lf, [256:384)=rf, [384:512)=u.
// Swizzle: flat = ((ntile*KCH + kc)*32 + lane)*16 + j, value = B[k][col] with
//   col = ntile*16 + (lane&15), k = kc*32 + (lane>>4)*16 + j.
// ---------------------------------------------------------------------------
__global__ void prep_kernel(const float* __restrict__ W_child, const float* __restrict__ b_child,
                            const float* __restrict__ W_x,     const float* __restrict__ b_x,
                            const float* __restrict__ W_leaf,  const float* __restrict__ b_leaf,
                            unsigned short* __restrict__ Blev, unsigned short* __restrict__ Bleaf,
                            float* __restrict__ biaslev,       float* __restrict__ biasleaf) {
  int tid = threadIdx.x;
  // Level B: 32 ntiles * 12 kchunks * 32 lanes * 16 j
  for (int idx = tid; idx < 32 * 12 * 32 * 16; idx += blockDim.x) {
    int j    = idx & 15;
    int lane = (idx >> 4) & 31;
    int ck   = idx >> 9;
    int kc   = ck % 12;
    int nt   = ck / 12;
    int col  = nt * 16 + (lane & 15);
    int kk   = kc * 32 + (lane >> 4) * 16 + j;
    int g    = col >> 7, lcol = col & 127;
    int seg  = kk >> 7,  l    = kk & 127;
    float v;
    if (seg == 0) {                       // xe @ W_x[gx]
      int gx = (g == 0) ? 0 : ((g == 3) ? 2 : 1);
      v = W_x[(gx * 128 + l) * 128 + lcol];
    } else if (seg == 1) {                // lh @ W_child[0::2][g] = W_child[2g]
      v = W_child[((2 * g) * 128 + l) * 128 + lcol];
    } else {                              // rh @ W_child[1::2][g] = W_child[2g+1]
      v = W_child[((2 * g + 1) * 128 + l) * 128 + lcol];
    }
    Blev[idx] = f2bf(v);
  }
  // Leaf B: 16 ntiles * 4 kchunks * 32 lanes * 16 j ; logical (128 x 256)
  for (int idx = tid; idx < 16 * 4 * 32 * 16; idx += blockDim.x) {
    int j    = idx & 15;
    int lane = (idx >> 4) & 31;
    int ck   = idx >> 9;
    int kc   = ck & 3;
    int nt   = ck >> 2;
    int col  = nt * 16 + (lane & 15);
    int kk   = kc * 32 + (lane >> 4) * 16 + j;
    int g    = col >> 7, lcol = col & 127;
    Bleaf[idx] = f2bf(W_leaf[(g * 128 + kk) * 128 + lcol]);
  }
  // Folded biases
  for (int col = tid; col < 512; col += blockDim.x) {
    int g = col >> 7, lcol = col & 127;
    int gx = (g == 0) ? 0 : ((g == 3) ? 2 : 1);
    biaslev[col] = b_x[gx * 128 + lcol] + b_child[(2 * g) * 128 + lcol] +
                   b_child[(2 * g + 1) * 128 + lcol];
  }
  for (int col = tid; col < 256; col += blockDim.x) {
    int g = col >> 7, lcol = col & 127;
    biasleaf[col] = b_leaf[g * 128 + lcol];
  }
}

// ---------------------------------------------------------------------------
// Leaf: 32 rows/block, A = gathered node_embedding row (bf16, LDS),
// B = packed (128 x 256). Wave w owns N-tiles {w, w+8}, M-tiles {0,1}.
// ---------------------------------------------------------------------------
__global__ void __launch_bounds__(256) leaf_kernel(
    const float* __restrict__ node_emb, const int* __restrict__ leaf_idx,
    const unsigned short* __restrict__ Bsw, const float* __restrict__ bias,
    float* __restrict__ c_buf, unsigned short* __restrict__ h_buf) {
  __shared__ unsigned short A[32 * 128];
  __shared__ int nodeIdx[32];
  int tid = threadIdx.x;
  int bid = blockIdx.x;
  int k  = bid >> 6;            // 64 row-tiles per tree
  int i0 = (bid & 63) * 32;
  int w = tid >> 5, lane = tid & 31;
  // Warm L2/WGP$ for this wave's packed-B stream (global_prefetch_b8).
  __builtin_prefetch(&Bsw[((w * 4) * 32 + lane) * 16], 0, 3);
  if (tid < 32) nodeIdx[tid] = leaf_idx[k * NLEAF + i0 + tid];
  __syncthreads();
  for (int idx = tid; idx < 32 * 128; idx += 256) {
    int r = idx >> 7, l = idx & 127;
    A[idx] = f2bf(node_emb[(size_t)nodeIdx[r] * 128 + l]);
  }
  __syncthreads();

  int half = lane >> 4, cl = lane & 15;
  v8f acc[2][2];
  for (int mt = 0; mt < 2; ++mt)
    for (int g = 0; g < 2; ++g)
      for (int i = 0; i < 8; ++i) acc[mt][g][i] = 0.0f;

#pragma unroll
  for (int kc = 0; kc < 4; ++kc) {
    FragBF a[2];
#pragma unroll
    for (int mt = 0; mt < 2; ++mt) {
      const uint4* p = (const uint4*)&A[(mt * 16 + cl) * 128 + kc * 32 + half * 16];
      a[mt].q[0] = p[0];          // K = kc*32 + half*16 + {0..7}
      a[mt].q[1] = p[1];          // K = kc*32 + half*16 + 8 + {0..7}
    }
#pragma unroll
    for (int g = 0; g < 2; ++g) {
      int nt = w + 8 * g;
      FragBF b;
      const uint4* p = (const uint4*)&Bsw[(((nt * 4) + kc) * 32 + lane) * 16];
      b.q[0] = p[0]; b.q[1] = p[1];
      acc[0][g] = wmma_bf16(a[0].v, b.v, acc[0][g]);
      acc[1][g] = wmma_bf16(a[1].v, b.v, acc[1][g]);
    }
  }

  int lcol = w * 16 + cl;
  float bc = bias[lcol], bh = bias[128 + lcol];
#pragma unroll
  for (int mt = 0; mt < 2; ++mt) {
#pragma unroll
    for (int v = 0; v < 8; ++v) {
      int i = i0 + mt * 16 + half * 8 + v;          // C/D: M = v + 8*half
      float c = acc[mt][0][v] + bc;
      float h = sigm(acc[mt][1][v] + bh) * tanhf(c);
      size_t o = ((size_t)(k * NLEAF + i)) * 128 + lcol;
      c_buf[o] = c;
      h_buf[o] = f2bf(h);
    }
  }
}

// ---------------------------------------------------------------------------
// Level d (d=0..5, m>=32): 32 output rows/block. A = [xe|lh|rh] (32x384 bf16).
// Wave w owns N-tiles {w, w+8, w+16, w+24} (gates i/lf/rf/u, same lcol range)
// and both M-tiles -> 8 accumulators, 96 WMMAs per wave.
// Node (k,j) of level d stored at slot j<<(d+1); children at j<<(d+1) (+ 1<<d).
// ---------------------------------------------------------------------------
__global__ void __launch_bounds__(256) level_kernel(
    const float* __restrict__ char_emb, const int* __restrict__ op_idx,
    const unsigned short* __restrict__ Bsw, const float* __restrict__ bias,
    float* __restrict__ c_buf, unsigned short* __restrict__ h_buf,
    int d, int m, int start, int tilesPerTree) {
  __shared__ unsigned short A[32 * 384];
  __shared__ int tokA[32], posLA[32], posRA[32];
  int tid = threadIdx.x;
  int bid = blockIdx.x;
  int k  = bid / tilesPerTree;
  int j0 = (bid % tilesPerTree) * 32;
  int w = tid >> 5, lane = tid & 31;
  __builtin_prefetch(&Bsw[((w * 12) * 32 + lane) * 16], 0, 3);

  if (tid < 32) {
    int jj = j0 + tid; if (jj > m - 1) jj = m - 1;
    tokA[tid]  = op_idx[k * 2047 + start + jj];
    int pl     = jj << (d + 1);
    posLA[tid] = pl;
    posRA[tid] = pl + (1 << d);
  }
  __syncthreads();

  size_t hbase = (size_t)k * NLEAF * 128;
  for (int idx = tid; idx < 32 * 128; idx += 256) {
    int r = idx >> 7, l = idx & 127;
    A[r * 384 + l]       = f2bf(char_emb[tokA[r] * 128 + l]);
    A[r * 384 + 128 + l] = h_buf[hbase + (size_t)posLA[r] * 128 + l];
    A[r * 384 + 256 + l] = h_buf[hbase + (size_t)posRA[r] * 128 + l];
  }
  __syncthreads();

  int half = lane >> 4, cl = lane & 15;
  v8f acc[2][4];
  for (int mt = 0; mt < 2; ++mt)
    for (int g = 0; g < 4; ++g)
      for (int i = 0; i < 8; ++i) acc[mt][g][i] = 0.0f;

#pragma unroll
  for (int kc = 0; kc < 12; ++kc) {
    FragBF a[2];
#pragma unroll
    for (int mt = 0; mt < 2; ++mt) {
      const uint4* p = (const uint4*)&A[(mt * 16 + cl) * 384 + kc * 32 + half * 16];
      a[mt].q[0] = p[0];
      a[mt].q[1] = p[1];
    }
#pragma unroll
    for (int g = 0; g < 4; ++g) {
      int nt = w + 8 * g;
      FragBF b;
      const uint4* p = (const uint4*)&Bsw[(((nt * 12) + kc) * 32 + lane) * 16];
      b.q[0] = p[0]; b.q[1] = p[1];
      acc[0][g] = wmma_bf16(a[0].v, b.v, acc[0][g]);
      acc[1][g] = wmma_bf16(a[1].v, b.v, acc[1][g]);
    }
  }

  int lcol = w * 16 + cl;
  float bi  = bias[lcol],       blf = bias[128 + lcol];
  float brf = bias[256 + lcol], bu  = bias[384 + lcol];
#pragma unroll
  for (int mt = 0; mt < 2; ++mt) {
#pragma unroll
    for (int v = 0; v < 8; ++v) {
      int jj = j0 + mt * 16 + half * 8 + v;
      if (jj < m) {
        int pl = jj << (d + 1);
        int pr = pl + (1 << d);
        size_t oL = hbase + (size_t)pl * 128 + lcol;
        size_t oR = hbase + (size_t)pr * 128 + lcol;
        float lc = c_buf[oL], rc = c_buf[oR];
        float gi  = sigm(acc[mt][0][v] + bi);
        float glf = sigm(acc[mt][1][v] + blf);
        float grf = sigm(acc[mt][2][v] + brf);
        float gu  = tanhf(acc[mt][3][v] + bu);
        float c = gi * gu + glf * lc + grf * rc;
        c_buf[oL] = c;                       // in-place: left-child slot
        h_buf[oL] = f2bf(tanhf(c));
      }
    }
  }
}

// ---------------------------------------------------------------------------
// Fused tail: levels d=6..10 (m=16,8,4,2,1). One block per tree; the tree's
// remaining 32 children (c: 16KB f32, h: 8KB bf16) live in LDS, compacted.
// Per level: stage A (<=16 rows), 48 WMMAs (one M-tile), gates, barriered
// in-place compaction in LDS. Root c written back to c_buf slot 0.
// ---------------------------------------------------------------------------
__global__ void __launch_bounds__(256) tail_kernel(
    const float* __restrict__ char_emb, const int* __restrict__ op_idx,
    const unsigned short* __restrict__ Bsw, const float* __restrict__ bias,
    float* __restrict__ c_buf, unsigned short* __restrict__ h_buf) {
  __shared__ float          cc[32 * 128];   // compact child c
  __shared__ unsigned short hh[32 * 128];   // compact child h (bf16)
  __shared__ unsigned short A[16 * 384];
  __shared__ int tokA[16];
  int tid = threadIdx.x;
  int k   = blockIdx.x;
  int w = tid >> 5, lane = tid & 31;
  int half = lane >> 4, cl = lane & 15;
  __builtin_prefetch(&Bsw[((w * 12) * 32 + lane) * 16], 0, 3);

  // Load level-6 children: 32 nodes at slots r<<6, compacted to rank r.
  size_t base = (size_t)k * NLEAF * 128;
  for (int idx = tid; idx < 32 * 128; idx += 256) {
    int r = idx >> 7, l = idx & 127;
    size_t o = base + ((size_t)(r << 6)) * 128 + l;
    cc[idx] = c_buf[o];
    hh[idx] = h_buf[o];
  }
  __syncthreads();

  int lcol = w * 16 + cl;
  float bi  = bias[lcol],       blf = bias[128 + lcol];
  float brf = bias[256 + lcol], bu  = bias[384 + lcol];

  for (int t = 0; t < 5; ++t) {
    int m     = 16 >> t;                 // outputs this level (d = 6+t)
    int startd = 2048 - (32 >> t);       // 2048 - (2048 >> d)
    // (a) stage op tokens
    if (tid < 16) {
      int jj = tid; if (jj > m - 1) jj = m - 1;
      tokA[tid] = op_idx[k * 2047 + startd + jj];
    }
    __syncthreads();
    // (c) stage A = [xe | lh | rh], 16 rows (rows >= m duplicate row m-1)
    for (int idx = tid; idx < 16 * 128; idx += 256) {
      int r = idx >> 7, l = idx & 127;
      int jr = (r > m - 1) ? (m - 1) : r;
      A[r * 384 + l]       = f2bf(char_emb[tokA[r] * 128 + l]);
      A[r * 384 + 128 + l] = hh[(2 * jr) * 128 + l];
      A[r * 384 + 256 + l] = hh[(2 * jr + 1) * 128 + l];
    }
    __syncthreads();
    // (e) WMMA: one M-tile, 4 gate N-tiles per wave
    v8f acc[4];
    for (int g = 0; g < 4; ++g)
      for (int i = 0; i < 8; ++i) acc[g][i] = 0.0f;
#pragma unroll
    for (int kc = 0; kc < 12; ++kc) {
      FragBF a;
      const uint4* p = (const uint4*)&A[cl * 384 + kc * 32 + half * 16];
      a.q[0] = p[0];
      a.q[1] = p[1];
#pragma unroll
      for (int g = 0; g < 4; ++g) {
        int nt = w + 8 * g;
        FragBF b;
        const uint4* pb = (const uint4*)&Bsw[(((nt * 12) + kc) * 32 + lane) * 16];
        b.q[0] = pb[0]; b.q[1] = pb[1];
        acc[g] = wmma_bf16(a.v, b.v, acc[g]);
      }
    }
    // gates: read children (into regs), barrier, write compacted outputs
    float cnew[8];
#pragma unroll
    for (int v = 0; v < 8; ++v) {
      int jj = half * 8 + v;
      if (jj < m) {
        float lc = cc[(2 * jj) * 128 + lcol];
        float rc = cc[(2 * jj + 1) * 128 + lcol];
        float gi  = sigm(acc[0][v] + bi);
        float glf = sigm(acc[1][v] + blf);
        float grf = sigm(acc[2][v] + brf);
        float gu  = tanhf(acc[3][v] + bu);
        cnew[v] = gi * gu + glf * lc + grf * rc;
      }
    }
    __syncthreads();                     // (f) all reads done before writes
#pragma unroll
    for (int v = 0; v < 8; ++v) {
      int jj = half * 8 + v;
      if (jj < m) {
        cc[jj * 128 + lcol] = cnew[v];
        hh[jj * 128 + lcol] = f2bf(tanhf(cnew[v]));
      }
    }
    __syncthreads();                     // (h) writes visible to next level
  }

  // Root c (compact row 0) -> c_buf slot 0 for the final stage.
  if (tid < 128) c_buf[base + tid] = cc[tid];
}

// ---------------------------------------------------------------------------
// Final stage in full fp32 (negligible work; precision matters here).
// h_root recomputed as tanh(c_root) from f32 c, so no bf16 touches this path.
// ---------------------------------------------------------------------------
__global__ void __launch_bounds__(256) final_kernel(
    const float* __restrict__ init_emb, const float* __restrict__ W_and,
    const float* __restrict__ b_and,    const float* __restrict__ W_oend,
    const float* __restrict__ b_oend,   const float* __restrict__ c_buf,
    float* __restrict__ out) {
  __shared__ float s[384];
  int tid = threadIdx.x;
  if (tid < 128) s[tid] = init_emb[tid];
  int dd = tid & 127;
  bool doH = tid >= 128;
  const float* Wm = doH ? (W_and + 16384) : W_and;
  float bb = doH ? b_and[128 + dd] : b_and[dd];
  float mn = 3.4e38f;
  for (int k = 0; k < NTREE; ++k) {
    const float* cr = c_buf + (size_t)k * NLEAF * 128;  // root at slot 0
    float acc = bb;
    for (int l = 0; l < 128; ++l) {
      float x = cr[l];
      if (doH) x = tanhf(x);
      acc += x * Wm[l * 128 + dd];
    }
    mn = fminf(mn, acc);
  }
  s[(doH ? 256 : 128) + dd] = mn;
  __syncthreads();
  if (tid < 128) {
    float acc = b_oend[tid];
    for (int j = 0; j < 384; ++j) acc += s[j] * W_oend[j * 128 + tid];
    out[tid] = tanhf(acc);
  }
}

// ---------------------------------------------------------------------------
extern "C" void kernel_launch(void* const* d_in, const int* in_sizes, int n_in,
                              void* d_out, int out_size, void* d_ws, size_t ws_size,
                              hipStream_t stream) {
  (void)in_sizes; (void)n_in; (void)out_size; (void)ws_size;
  const float* node_emb = (const float*)d_in[0];
  const float* init_emb = (const float*)d_in[1];
  const float* char_emb = (const float*)d_in[2];
  const float* W_child  = (const float*)d_in[3];
  const float* b_child  = (const float*)d_in[4];
  const float* W_x      = (const float*)d_in[5];
  const float* b_x      = (const float*)d_in[6];
  const float* W_leaf   = (const float*)d_in[7];
  const float* b_leaf   = (const float*)d_in[8];
  const float* W_and    = (const float*)d_in[9];
  const float* b_and    = (const float*)d_in[10];
  const float* W_oend   = (const float*)d_in[11];
  const float* b_oend   = (const float*)d_in[12];
  const int*   leaf_idx = (const int*)d_in[13];
  const int*   op_idx   = (const int*)d_in[14];

  // Workspace layout (bytes), ~101.2 MB total:
  char* ws = (char*)d_ws;
  float*          c_buf    = (float*)ws;                          // 67,108,864
  unsigned short* h_buf    = (unsigned short*)(ws + 67108864);    // 33,554,432
  unsigned short* Blev     = (unsigned short*)(ws + 100663296);   //    393,216
  unsigned short* Bleaf    = (unsigned short*)(ws + 101056512);   //     65,536
  float*          biaslev  = (float*)(ws + 101122048);            //      2,048
  float*          biasleaf = (float*)(ws + 101124096);            //      1,024

  prep_kernel<<<1, 256, 0, stream>>>(W_child, b_child, W_x, b_x, W_leaf, b_leaf,
                                     Blev, Bleaf, biaslev, biasleaf);
  leaf_kernel<<<4096, 256, 0, stream>>>(node_emb, leaf_idx, Bleaf, biasleaf,
                                        c_buf, h_buf);
  int start = 0;
  for (int d = 0; d < 6; ++d) {          // m = 1024, 512, 256, 128, 64, 32
    int m = 1024 >> d;
    int tiles = m / 32;
    level_kernel<<<NTREE * tiles, 256, 0, stream>>>(char_emb, op_idx, Blev, biaslev,
                                                    c_buf, h_buf, d, m, start, tiles);
    start += m;
  }
  tail_kernel<<<NTREE, 256, 0, stream>>>(char_emb, op_idx, Blev, biaslev,
                                         c_buf, h_buf);
  final_kernel<<<1, 256, 0, stream>>>(init_emb, W_and, b_and, W_oend, b_oend,
                                      c_buf, (float*)d_out);
}
